// create_knowledge_84696755077745
// MI455X (gfx1250) — matrix operand order repaired and checked by reading, not compiled, not run on hardware
//
#include <hip/hip_runtime.h>
#include <hip/hip_bf16.h>

// ---------------------------------------------------------------------------
// Types for WMMA (CDNA5 / gfx1250, wave32)
// ---------------------------------------------------------------------------
typedef _Float16 half8 __attribute__((ext_vector_type(8)));
typedef _Float16 v16h  __attribute__((ext_vector_type(16)));
typedef float    v8f   __attribute__((ext_vector_type(8)));

// Problem constants (from reference)
constexpr int cBS = 16, cE = 256, cQ = 57600, cL = 90, cD = 768;
constexpr int cH = 6, cHD = 128, cK = 16, cNP = 197;
constexpr int cENC_FF = 3072, cDEC_FF = 1024, cNL = 2;
constexpr int T1 = cBS * cK * cL;   // 23040 encoder tokens
constexpr int T2 = cBS * cNP;       // 3152 decoder tokens
constexpr int TF = cBS * cL;        // 1440 "final" tokens

// ---------------------------------------------------------------------------
// sim = image_feat @ text_queue (pure f32: top-k selection is rounding sensitive)
// ---------------------------------------------------------------------------
__global__ void k_sim(const float* __restrict__ img, const float* __restrict__ tq,
                      float* __restrict__ sim) {
  int idx = blockIdx.x * blockDim.x + threadIdx.x;
  if (idx >= cBS * cQ) return;
  int b = idx / cQ, qc = idx % cQ;
  const float* ir = img + b * cE;
  float acc = 0.f;
  for (int e = 0; e < cE; ++e) acc += ir[e] * tq[(size_t)e * cQ + qc];
  sim[idx] = acc;
}

// ---------------------------------------------------------------------------
// top-16 (sorted desc, ties -> lower index) + softmax weights. One block/row.
// ---------------------------------------------------------------------------
__global__ void k_topk(const float* __restrict__ sim, int* __restrict__ tidx,
                       float* __restrict__ wsel) {
  int row = blockIdx.x, tid = threadIdx.x;
  __shared__ float sv[256];
  __shared__ int   si[256];
  __shared__ int   sel[cK];
  __shared__ float val[cK];
  const float* sr = sim + (size_t)row * cQ;
  for (int p = 0; p < cK; ++p) {
    float best = -INFINITY; int bi = 0x7fffffff;
    for (int i = tid; i < cQ; i += 256) {
      bool taken = false;
      for (int j = 0; j < p; ++j) if (sel[j] == i) taken = true;
      if (taken) continue;
      float v = sr[i];
      if (v > best || (v == best && i < bi)) { best = v; bi = i; }
    }
    sv[tid] = best; si[tid] = bi; __syncthreads();
    for (int s = 128; s > 0; s >>= 1) {
      if (tid < s) {
        float vo = sv[tid + s]; int io = si[tid + s];
        if (vo > sv[tid] || (vo == sv[tid] && io < si[tid])) { sv[tid] = vo; si[tid] = io; }
      }
      __syncthreads();
    }
    if (tid == 0) { sel[p] = si[0]; val[p] = sv[0]; }
    __syncthreads();
  }
  if (tid == 0) {
    float m = val[0], s = 0.f, e[cK];
    for (int j = 0; j < cK; ++j) { e[j] = expf(val[j] - m); s += e[j]; }
    for (int j = 0; j < cK; ++j) { wsel[row * cK + j] = e[j] / s; tidx[row * cK + j] = sel[j]; }
  }
}

// ---------------------------------------------------------------------------
// Transpose-convert weight: src[K,N] f32 row-major -> dst[N,K] f16 row-major
// ---------------------------------------------------------------------------
__global__ void k_tconv(_Float16* __restrict__ dst, const float* __restrict__ src,
                        int K, int N) {
  size_t total = (size_t)K * N;
  for (size_t i = blockIdx.x * (size_t)blockDim.x + threadIdx.x; i < total;
       i += (size_t)gridDim.x * blockDim.x) {
    size_t n = i / K, k = i % K;
    dst[i] = (_Float16)src[k * (size_t)N + n];
  }
}

// ---------------------------------------------------------------------------
// Gather retrieved sequences: h = emb_table[ids] + pos_emb ; gather mask
// grid (cL, 256)
// ---------------------------------------------------------------------------
__global__ void k_gather(const int* __restrict__ tidx, const int* __restrict__ idsq,
                         const float* __restrict__ maskq, const float* __restrict__ emb,
                         const float* __restrict__ pos, float* __restrict__ h32,
                         _Float16* __restrict__ h16, float* __restrict__ msk) {
  int l = blockIdx.x, seq = blockIdx.y, tid = threadIdx.x;
  int b = seq / cK, ki = seq % cK;
  int qi = tidx[b * cK + ki];
  int id = idsq[(size_t)qi * cL + l];
  const float* er = emb + (size_t)id * cD;
  const float* pr = pos + (size_t)l * cD;
  size_t o = ((size_t)(seq * cL + l)) * cD;
  for (int d = tid; d < cD; d += blockDim.x) {
    float v = er[d] + pr[d];
    h32[o + d] = v; h16[o + d] = (_Float16)v;
  }
  if (tid == 0) msk[seq * cL + l] = maskq[(size_t)qi * cL + l];
}

// ---------------------------------------------------------------------------
// WMMA GEMM: C[M,N] = A[M,K](f16,row-major) @ Bt[N,K](f16,row-major == B^T)
// Each wave owns a 32x64 tile (8 accumulators). All 12 b128 loads of a k-step
// (2xA, 2xA' when FULL, 8xB in distinct registers) are issued before the WMMA
// burst so the scheduler can use partial s_wait_loadcnt instead of full drains.
// M%16==0, N%64==0, K%32==0. A 16-row remainder supertile (e.g. M=3152) takes
// the FULL=false path; the split is wave-uniform so EXEC is all-ones for WMMA.
// EPI: 0 = f32 store, 1 = f16 store, 2 = gelu(tanh)->f16, 3 = relu->f16
// ---------------------------------------------------------------------------
union GemmU8 { v8f v; float f[8]; };
union GemmAU { half8 h[2]; v16h v; };

template <int EPI>
__device__ __forceinline__ void gemm_store(void* __restrict__ Cp, float v, size_t off) {
  if (EPI == 0) ((float*)Cp)[off] = v;
  else if (EPI == 1) ((_Float16*)Cp)[off] = (_Float16)v;
  else if (EPI == 2) {
    float g = 0.5f * v * (1.f + tanhf(0.7978845608028654f * (v + 0.044715f * v * v * v)));
    ((_Float16*)Cp)[off] = (_Float16)g;
  } else ((_Float16*)Cp)[off] = (_Float16)(v > 0.f ? v : 0.f);
}

template <int EPI, bool FULL>
__device__ __forceinline__ void gemm_tile(const _Float16* __restrict__ A,
                                          const _Float16* __restrict__ Bt,
                                          void* __restrict__ Cp,
                                          int N, int K, int m0, int n0, int lane) {
  GemmU8 acc[2][4];
#pragma unroll
  for (int t = 0; t < 2; ++t)
#pragma unroll
    for (int j = 0; j < 4; ++j)
#pragma unroll
      for (int r = 0; r < 8; ++r) acc[t][j].f[r] = 0.f;

  // 16-bit A 16x32 layout (matches B): lane m (0-15): K 0..15; lane m+16: K 16..31
  int ar = lane & 15, ak = (lane >> 4) << 4;
  // 16-bit B 32x16 layout (matches A): lane n (0-15): K 0..15 of col n; lane n+16: K 16..31
  int bn = lane & 15, bk = (lane >> 4) << 4;
  const _Float16* ap0 = A + (size_t)(m0 + ar) * K + ak;
  const _Float16* ap1 = ap0 + (size_t)16 * K;
  const _Float16* bp0 = Bt + (size_t)(n0 + 0 * 16 + bn) * K + bk;
  const _Float16* bp1 = Bt + (size_t)(n0 + 1 * 16 + bn) * K + bk;
  const _Float16* bp2 = Bt + (size_t)(n0 + 2 * 16 + bn) * K + bk;
  const _Float16* bp3 = Bt + (size_t)(n0 + 3 * 16 + bn) * K + bk;

  for (int k0 = 0; k0 < K; k0 += 32) {
    // Issue all loads of this k-step first (distinct destination registers).
    GemmAU a0, a1, b0, b1, b2, b3;
    a0.h[0] = *(const half8*)(ap0 + k0);
    a0.h[1] = *(const half8*)(ap0 + k0 + 8);
    if (FULL) {
      a1.h[0] = *(const half8*)(ap1 + k0);
      a1.h[1] = *(const half8*)(ap1 + k0 + 8);
    }
    b0.h[0] = *(const half8*)(bp0 + k0); b0.h[1] = *(const half8*)(bp0 + k0 + 8);
    b1.h[0] = *(const half8*)(bp1 + k0); b1.h[1] = *(const half8*)(bp1 + k0 + 8);
    b2.h[0] = *(const half8*)(bp2 + k0); b2.h[1] = *(const half8*)(bp2 + k0 + 8);
    b3.h[0] = *(const half8*)(bp3 + k0); b3.h[1] = *(const half8*)(bp3 + k0 + 8);
    // WMMA burst (B fragments reused across both 16-row halves).
    acc[0][0].v = __builtin_amdgcn_wmma_f32_16x16x32_f16(false, a0.v, false, b0.v, (short)0, acc[0][0].v, false, false);
    if (FULL)
      acc[1][0].v = __builtin_amdgcn_wmma_f32_16x16x32_f16(false, a1.v, false, b0.v, (short)0, acc[1][0].v, false, false);
    acc[0][1].v = __builtin_amdgcn_wmma_f32_16x16x32_f16(false, a0.v, false, b1.v, (short)0, acc[0][1].v, false, false);
    if (FULL)
      acc[1][1].v = __builtin_amdgcn_wmma_f32_16x16x32_f16(false, a1.v, false, b1.v, (short)0, acc[1][1].v, false, false);
    acc[0][2].v = __builtin_amdgcn_wmma_f32_16x16x32_f16(false, a0.v, false, b2.v, (short)0, acc[0][2].v, false, false);
    if (FULL)
      acc[1][2].v = __builtin_amdgcn_wmma_f32_16x16x32_f16(false, a1.v, false, b2.v, (short)0, acc[1][2].v, false, false);
    acc[0][3].v = __builtin_amdgcn_wmma_f32_16x16x32_f16(false, a0.v, false, b3.v, (short)0, acc[0][3].v, false, false);
    if (FULL)
      acc[1][3].v = __builtin_amdgcn_wmma_f32_16x16x32_f16(false, a1.v, false, b3.v, (short)0, acc[1][3].v, false, false);
  }
  // D layout: VGPR r -> row m0 + r + (lane>=16 ? 8 : 0), col n0 + (lane&15)
  int cn = lane & 15, cm = (lane >> 4) << 3;
#pragma unroll
  for (int j = 0; j < 4; ++j)
#pragma unroll
    for (int r = 0; r < 8; ++r) {
      size_t off = (size_t)(m0 + cm + r) * N + n0 + j * 16 + cn;
      gemm_store<EPI>(Cp, acc[0][j].f[r], off);
      if (FULL) gemm_store<EPI>(Cp, acc[1][j].f[r], off + (size_t)16 * N);
    }
}

template <int EPI>
__global__ void k_gemm(const _Float16* __restrict__ A, const _Float16* __restrict__ Bt,
                       void* __restrict__ Cp, int M, int N, int K) {
  int wave = (int)((blockIdx.x * (size_t)blockDim.x + threadIdx.x) >> 5);
  int lane = threadIdx.x & 31;
  int ng4 = N >> 6;
  int mst = (M + 31) >> 5;            // 32-row supertiles (last may be 16 rows)
  if (wave >= mst * ng4) return;      // wave-uniform
  int m0 = (wave / ng4) << 5;
  int n0 = (wave % ng4) << 6;
  if (m0 + 32 <= M) gemm_tile<EPI, true >(A, Bt, Cp, N, K, m0, n0, lane);
  else              gemm_tile<EPI, false>(A, Bt, Cp, N, K, m0, n0, lane);
}

// ---------------------------------------------------------------------------
// Attention: out = softmax(q k^T / sqrt(HD) [mask]) v  per (seq, head, q-row)
// q/k/v/out are [tokens, 768] f16 with head h in cols [h*128, h*128+128).
// grid (Lq, H, nseq), block 128. mask may be null; mask[s*Lk + lk].
// ---------------------------------------------------------------------------
__global__ void k_attn(const _Float16* __restrict__ q, const _Float16* __restrict__ kk,
                       const _Float16* __restrict__ vv, _Float16* __restrict__ out,
                       const float* __restrict__ mask, int Lq, int Lk) {
  int lq = blockIdx.x, h = blockIdx.y, s = blockIdx.z, tid = threadIdx.x;
  __shared__ float qs[cHD];
  __shared__ float sc[256];
  __shared__ float red[128];
  const _Float16* qrow = q + ((size_t)s * Lq + lq) * cD + h * cHD;
  qs[tid] = (float)qrow[tid];
  __syncthreads();
  const float scale = 0.088388347648318447f;  // 1/sqrt(128)
  for (int lk = tid; lk < Lk; lk += 128) {
    const _Float16* krow = kk + ((size_t)s * Lk + lk) * cD + h * cHD;
    float d = 0.f;
#pragma unroll 8
    for (int e = 0; e < cHD; ++e) d += qs[e] * (float)krow[e];
    d *= scale;
    if (mask && !(mask[s * Lk + lk] > 0.f)) d = -1e9f;
    sc[lk] = d;
  }
  __syncthreads();
  float m = -INFINITY;
  for (int lk = tid; lk < Lk; lk += 128) m = fmaxf(m, sc[lk]);
  red[tid] = m; __syncthreads();
  for (int st = 64; st > 0; st >>= 1) { if (tid < st) red[tid] = fmaxf(red[tid], red[tid + st]); __syncthreads(); }
  m = red[0]; __syncthreads();
  float ps = 0.f;
  for (int lk = tid; lk < Lk; lk += 128) { float e = expf(sc[lk] - m); sc[lk] = e; ps += e; }
  red[tid] = ps; __syncthreads();
  for (int st = 64; st > 0; st >>= 1) { if (tid < st) red[tid] += red[tid + st]; __syncthreads(); }
  float inv = 1.f / red[0];
  float o = 0.f;
  for (int lk = 0; lk < Lk; ++lk)
    o += sc[lk] * (float)vv[((size_t)s * Lk + lk) * cD + h * cHD + tid];
  out[((size_t)s * Lq + lq) * cD + h * cHD + tid] = (_Float16)(o * inv);
}

// ---------------------------------------------------------------------------
// (x + y) [optional] -> LN [optional] -> dst32/dst16 (either may be null). D=768,
// block 256, one block per row. In-place safe (each elem read by its writer).
// ---------------------------------------------------------------------------
__global__ void k_addln(float* __restrict__ dst32, _Float16* __restrict__ dst16,
                        const float* __restrict__ x, const float* __restrict__ y,
                        int doLN) {
  int row = blockIdx.x, tid = threadIdx.x;
  __shared__ float red[256];
  size_t o = (size_t)row * cD;
  float v[3];
#pragma unroll
  for (int i = 0; i < 3; ++i) {
    int d = tid + i * 256;
    float t = x[o + d];
    if (y) t += y[o + d];
    v[i] = t;
  }
  if (doLN) {
    red[tid] = v[0] + v[1] + v[2]; __syncthreads();
    for (int st = 128; st > 0; st >>= 1) { if (tid < st) red[tid] += red[tid + st]; __syncthreads(); }
    float mean = red[0] / (float)cD; __syncthreads();
    float qs = 0.f;
#pragma unroll
    for (int i = 0; i < 3; ++i) { float dl = v[i] - mean; qs += dl * dl; }
    red[tid] = qs; __syncthreads();
    for (int st = 128; st > 0; st >>= 1) { if (tid < st) red[tid] += red[tid + st]; __syncthreads(); }
    float var = red[0] / (float)cD;
    float r = rsqrtf(var + 1e-6f);
#pragma unroll
    for (int i = 0; i < 3; ++i) v[i] = (v[i] - mean) * r;
  }
#pragma unroll
  for (int i = 0; i < 3; ++i) {
    int d = tid + i * 256;
    if (dst32) dst32[o + d] = v[i];
    if (dst16) dst16[o + d] = (_Float16)v[i];
  }
}

__global__ void k_add(float* __restrict__ x, const float* __restrict__ y, size_t n) {
  for (size_t i = blockIdx.x * (size_t)blockDim.x + threadIdx.x; i < n;
       i += (size_t)gridDim.x * blockDim.x) x[i] += y[i];
}

// final[b,l,:] = sum_k w[b,k] * h[(b*16+k)*90 + l, :]   grid (cL, cBS)
__global__ void k_wfinal(const float* __restrict__ h, const float* __restrict__ wsel,
                         float* __restrict__ f32o, _Float16* __restrict__ f16o) {
  int l = blockIdx.x, b = blockIdx.y, tid = threadIdx.x;
  for (int d = tid; d < cD; d += blockDim.x) {
    float acc = 0.f;
    for (int ki = 0; ki < cK; ++ki)
      acc += wsel[b * cK + ki] * h[((size_t)((b * cK + ki) * cL + l)) * cD + d];
    size_t o = ((size_t)(b * cL + l)) * cD + d;
    f32o[o] = acc; f16o[o] = (_Float16)acc;
  }
}

// ---------------------------------------------------------------------------
// Host orchestration
// ---------------------------------------------------------------------------
extern "C" void kernel_launch(void* const* d_in, const int* in_sizes, int n_in,
                              void* d_out, int out_size, void* d_ws, size_t ws_size,
                              hipStream_t stream) {
  (void)in_sizes; (void)n_in; (void)out_size;
  const float* image_feat   = (const float*)d_in[0];
  const float* image_embeds = (const float*)d_in[1];
  const float* text_queue   = (const float*)d_in[2];
  const float* mask_queue   = (const float*)d_in[3];
  const float* emb_table    = (const float*)d_in[4];
  const float* pos_emb      = (const float*)d_in[5];
  const float* enc_wq = (const float*)d_in[6];
  const float* enc_wk = (const float*)d_in[7];
  const float* enc_wv = (const float*)d_in[8];
  const float* enc_wo = (const float*)d_in[9];
  const float* enc_w1 = (const float*)d_in[10];
  const float* enc_w2 = (const float*)d_in[11];
  const float* dec_self_w  = (const float*)d_in[12];
  const float* dec_cross_w = (const float*)d_in[13];
  const float* dec_w1 = (const float*)d_in[14];
  const float* dec_w2 = (const float*)d_in[15];
  const int*   ids_queue = (const int*)d_in[16];
  // d_in[17] = k (hardcoded 16)

  char* w = (char*)d_ws;
  size_t cur = 0;
  auto alloc = [&](size_t bytes) -> char* {
    char* p = w + cur; cur = (cur + bytes + 255) & ~(size_t)255; return p;
  };

  float* sim  = (float*)alloc((size_t)cBS * cQ * 4);
  int*   tidx = (int*)alloc(cBS * cK * 4);
  float* wsel = (float*)alloc(cBS * cK * 4);
  float* msk  = (float*)alloc((size_t)cBS * cK * cL * 4);

  _Float16 *wq_t = (_Float16*)alloc((size_t)cD * cD * 2);
  _Float16 *wk_t = (_Float16*)alloc((size_t)cD * cD * 2);
  _Float16 *wv_t = (_Float16*)alloc((size_t)cD * cD * 2);
  _Float16 *wo_t = (_Float16*)alloc((size_t)cD * cD * 2);
  _Float16 *w1_t = (_Float16*)alloc((size_t)cD * cENC_FF * 2);
  _Float16 *w2_t = (_Float16*)alloc((size_t)cENC_FF * cD * 2);
  _Float16 *dswt[cNL][4], *dcwt[cNL][4], *dw1t[cNL], *dw2t[cNL];
  for (int l = 0; l < cNL; ++l) {
    for (int j = 0; j < 4; ++j) dswt[l][j] = (_Float16*)alloc((size_t)cD * cD * 2);
    for (int j = 0; j < 4; ++j) dcwt[l][j] = (_Float16*)alloc((size_t)cD * cD * 2);
    dw1t[l] = (_Float16*)alloc((size_t)cD * cDEC_FF * 2);
    dw2t[l] = (_Float16*)alloc((size_t)cDEC_FF * cD * 2);
  }
  float*     final32 = (float*)alloc((size_t)TF * cD * 4);
  _Float16*  final16 = (_Float16*)alloc((size_t)TF * cD * 2);

  // Encoder region (reused by decoder after `final` is computed)
  size_t encMark = cur;
  float*    h32   = (float*)alloc((size_t)T1 * cD * 4);
  float*    tmp32 = (float*)alloc((size_t)T1 * cD * 4);
  _Float16* h16   = (_Float16*)alloc((size_t)T1 * cD * 2);
  _Float16* q16   = (_Float16*)alloc((size_t)T1 * cD * 2);
  _Float16* k16   = (_Float16*)alloc((size_t)T1 * cD * 2);
  _Float16* v16   = (_Float16*)alloc((size_t)T1 * cD * 2);
  _Float16* ao16  = (_Float16*)alloc((size_t)T1 * cD * 2);
  _Float16* g16   = q16;  // alias: q/k/v/ao (4*T1*D) == T1*ENC_FF halves; safe by ordering
  size_t need = cur;
  if (ws_size < need) return;  // workspace too small: bail (deterministic no-op)

  // Decoder buffers alias the encoder region (used only after k_wfinal)
  size_t dcur = encMark;
  auto dalloc = [&](size_t bytes) -> char* {
    char* p = w + dcur; dcur = (dcur + bytes + 255) & ~(size_t)255; return p;
  };
  float*    x32  = (float*)dalloc((size_t)T2 * cD * 4);
  float*    t2a  = (float*)dalloc((size_t)T2 * cD * 4);
  _Float16* xn16 = (_Float16*)dalloc((size_t)T2 * cD * 2);
  _Float16* qd16 = (_Float16*)dalloc((size_t)T2 * cD * 2);
  _Float16* kd16 = (_Float16*)dalloc((size_t)T2 * cD * 2);
  _Float16* vd16 = (_Float16*)dalloc((size_t)T2 * cD * 2);
  _Float16* aod16= (_Float16*)dalloc((size_t)T2 * cD * 2);
  _Float16* kc16 = (_Float16*)dalloc((size_t)TF * cD * 2);
  _Float16* vc16 = (_Float16*)dalloc((size_t)TF * cD * 2);
  _Float16* fd16 = (_Float16*)dalloc((size_t)T2 * cDEC_FF * 2);

  auto tcv = [&](_Float16* dst, const float* src, int K, int N) {
    size_t total = (size_t)K * N;
    int blocks = (int)((total + 255) / 256); if (blocks > 8192) blocks = 8192;
    k_tconv<<<blocks, 256, 0, stream>>>(dst, src, K, N);
  };
  auto gemm = [&](int epi, const _Float16* A, const _Float16* Bt, void* C,
                  int M, int N, int K) {
    int waves = ((M + 31) / 32) * (N / 64);
    int blocks = (waves + 7) / 8;
    switch (epi) {
      case 0: k_gemm<0><<<blocks, 256, 0, stream>>>(A, Bt, C, M, N, K); break;
      case 1: k_gemm<1><<<blocks, 256, 0, stream>>>(A, Bt, C, M, N, K); break;
      case 2: k_gemm<2><<<blocks, 256, 0, stream>>>(A, Bt, C, M, N, K); break;
      default: k_gemm<3><<<blocks, 256, 0, stream>>>(A, Bt, C, M, N, K); break;
    }
  };

  // ---- weight transposition/conversion (stateless: every call) ----
  tcv(wq_t, enc_wq, cD, cD); tcv(wk_t, enc_wk, cD, cD);
  tcv(wv_t, enc_wv, cD, cD); tcv(wo_t, enc_wo, cD, cD);
  tcv(w1_t, enc_w1, cD, cENC_FF); tcv(w2_t, enc_w2, cENC_FF, cD);
  for (int l = 0; l < cNL; ++l) {
    for (int j = 0; j < 4; ++j) {
      tcv(dswt[l][j], dec_self_w  + (size_t)(l * 4 + j) * cD * cD, cD, cD);
      tcv(dcwt[l][j], dec_cross_w + (size_t)(l * 4 + j) * cD * cD, cD, cD);
    }
    tcv(dw1t[l], dec_w1 + (size_t)l * cD * cDEC_FF, cD, cDEC_FF);
    tcv(dw2t[l], dec_w2 + (size_t)l * cDEC_FF * cD, cDEC_FF, cD);
  }

  // ---- retrieval ----
  k_sim<<<(cBS * cQ + 255) / 256, 256, 0, stream>>>(image_feat, text_queue, sim);
  k_topk<<<cBS, 256, 0, stream>>>(sim, tidx, wsel);
  k_gather<<<dim3(cL, cBS * cK), 256, 0, stream>>>(tidx, ids_queue, mask_queue,
                                                   emb_table, pos_emb, h32, h16, msk);

  // ---- knowledge encoder (post-LN BERT layer) ----
  gemm(1, h16, wq_t, q16, T1, cD, cD);
  gemm(1, h16, wk_t, k16, T1, cD, cD);
  gemm(1, h16, wv_t, v16, T1, cD, cD);
  k_attn<<<dim3(cL, cH, cBS * cK), 128, 0, stream>>>(q16, k16, v16, ao16, msk, cL, cL);
  gemm(0, ao16, wo_t, tmp32, T1, cD, cD);
  k_addln<<<T1, 256, 0, stream>>>(h32, h16, h32, tmp32, 1);       // h = LN(h + attn)
  gemm(2, h16, w1_t, g16, T1, cENC_FF, cD);                       // gelu fused -> f16
  gemm(0, g16, w2_t, tmp32, T1, cD, cENC_FF);
  k_addln<<<T1, 256, 0, stream>>>(h32, nullptr, h32, tmp32, 1);   // h = LN(h + ff)
  k_wfinal<<<dim3(cL, cBS), 256, 0, stream>>>(h32, wsel, final32, final16);

  // ---- cross-attention decoder (pre-LN) ----
  hipMemcpyAsync(x32, image_embeds, (size_t)T2 * cD * 4, hipMemcpyDeviceToDevice, stream);
  for (int l = 0; l < cNL; ++l) {
    // self-attention
    k_addln<<<T2, 256, 0, stream>>>(nullptr, xn16, x32, nullptr, 1);
    gemm(1, xn16, dswt[l][0], qd16, T2, cD, cD);
    gemm(1, xn16, dswt[l][1], kd16, T2, cD, cD);
    gemm(1, xn16, dswt[l][2], vd16, T2, cD, cD);
    k_attn<<<dim3(cNP, cH, cBS), 128, 0, stream>>>(qd16, kd16, vd16, aod16, nullptr, cNP, cNP);
    gemm(0, aod16, dswt[l][3], t2a, T2, cD, cD);
    k_add<<<(int)(((size_t)T2 * cD + 255) / 256), 256, 0, stream>>>(x32, t2a, (size_t)T2 * cD);
    // cross-attention
    k_addln<<<T2, 256, 0, stream>>>(nullptr, xn16, x32, nullptr, 1);
    gemm(1, xn16, dcwt[l][0], qd16, T2, cD, cD);
    gemm(1, final16, dcwt[l][1], kc16, TF, cD, cD);
    gemm(1, final16, dcwt[l][2], vc16, TF, cD, cD);
    k_attn<<<dim3(cNP, cH, cBS), 128, 0, stream>>>(qd16, kc16, vc16, aod16, nullptr, cNP, cL);
    gemm(0, aod16, dcwt[l][3], t2a, T2, cD, cD);
    k_add<<<(int)(((size_t)T2 * cD + 255) / 256), 256, 0, stream>>>(x32, t2a, (size_t)T2 * cD);
    // feed-forward (relu)
    k_addln<<<T2, 256, 0, stream>>>(nullptr, xn16, x32, nullptr, 1);
    gemm(3, xn16, dw1t[l], fd16, T2, cDEC_FF, cD);                // relu fused -> f16
    gemm(0, fd16, dw2t[l], t2a, T2, cD, cDEC_FF);
    k_add<<<(int)(((size_t)T2 * cD + 255) / 256), 256, 0, stream>>>(x32, t2a, (size_t)T2 * cD);
  }
  k_addln<<<T2, 256, 0, stream>>>((float*)d_out, nullptr, x32, nullptr, 1);
}
